// BayesRNNBase_31181462569110
// MI455X (gfx1250) — matrix operand order repaired and checked by reading, not compiled, not run on hardware
//
#include <hip/hip_runtime.h>
#include <hip/hip_bf16.h>
#include <math.h>

// ---------------- types ----------------
typedef __attribute__((ext_vector_type(16))) __bf16     v16bf;
typedef __attribute__((ext_vector_type(8)))  float      v8f;
typedef __attribute__((ext_vector_type(4)))  unsigned   u32x4;

union ABv { u32x4 u[2]; v16bf v; };   // 32 bytes: two b128 loads -> one WMMA operand

#define TT 256
#define BB 128
#define II 256
#define HH 1024
#define GG (4*HH)

__device__ __forceinline__ float sigmoidf_(float x) { return 1.0f / (1.0f + __expf(-x)); }

// ---------------- init: zero h0 (bf16), c0 (f32) ----------------
__global__ __launch_bounds__(256) void init_state_kernel(float* c, __bf16* h0, int n) {
    int i = blockIdx.x * 256 + threadIdx.x;
    if (i < n) { c[i] = 0.0f; h0[i] = (__bf16)0.0f; }
}

// ---------------- fp32 -> bf16 convert (x tensor); x read once -> NT load ----------------
__global__ __launch_bounds__(256) void f32_to_bf16_kernel(const float* __restrict__ in,
                                                          __bf16* __restrict__ out, int n) {
    int i = blockIdx.x * 256 + threadIdx.x;
    if (i < n) out[i] = (__bf16)__builtin_nontemporal_load(&in[i]);
}

// ---------------- sample weights + per-block KL partials ----------------
// partial[blk*3 + {0,1,2}] = { sum log_post, sum logpdf(w,0,1), sum logpdf(w,0,1e-3) }
__global__ __launch_bounds__(256) void sample_kl_kernel(const float* __restrict__ mu,
                                                        const float* __restrict__ rho,
                                                        const float* __restrict__ eps,
                                                        __bf16* __restrict__ wout,
                                                        float* __restrict__ partial, int n) {
    const float HL2PI = 0.91893853320467274f;   // 0.5*log(2*pi)
    int i = blockIdx.x * 256 + threadIdx.x;
    float lp = 0.f, m1 = 0.f, m2 = 0.f;
    if (i < n) {
        float r = __builtin_nontemporal_load(&rho[i]);
        float s = ((r > 15.f) ? r : log1pf(__expf(r))) + 1e-5f;
        float e = __builtin_nontemporal_load(&eps[i]);
        float w = __builtin_nontemporal_load(&mu[i]) + e * s;
        wout[i] = (__bf16)w;                            // re-read every step: keep in L2
        lp = -0.5f * e * e - logf(s) - HL2PI;           // z = (w-mu)/s == eps exactly
        m1 = -0.5f * w * w - HL2PI;                     // sigma1 = 1
        float z2 = w * 1000.0f;                         // sigma2 = 1e-3
        m2 = -0.5f * z2 * z2 - logf(0.001f) - HL2PI;
    }
    // deterministic wave tree-reduce, then cross-wave via LDS
    for (int o = 16; o > 0; o >>= 1) {
        lp += __shfl_down(lp, o); m1 += __shfl_down(m1, o); m2 += __shfl_down(m2, o);
    }
    __shared__ float sw[8][3];
    int wave = threadIdx.x >> 5;
    if ((threadIdx.x & 31) == 0) { sw[wave][0] = lp; sw[wave][1] = m1; sw[wave][2] = m2; }
    __syncthreads();
    if (threadIdx.x == 0) {
        float a = 0, b = 0, c = 0;
        #pragma unroll
        for (int w = 0; w < 8; ++w) { a += sw[w][0]; b += sw[w][1]; c += sw[w][2]; }
        partial[blockIdx.x * 3 + 0] = a;
        partial[blockIdx.x * 3 + 1] = b;
        partial[blockIdx.x * 3 + 2] = c;
    }
}

// ---------------- sample both biases, bc = b_ih + b_hh, 6 KL partials/blk ----------------
__global__ __launch_bounds__(256) void sample_bias_kernel(const float* __restrict__ mu1,
                                                          const float* __restrict__ rho1,
                                                          const float* __restrict__ e1,
                                                          const float* __restrict__ mu2,
                                                          const float* __restrict__ rho2,
                                                          const float* __restrict__ e2,
                                                          float* __restrict__ bc,
                                                          float* __restrict__ partial, int n) {
    const float HL2PI = 0.91893853320467274f;
    int i = blockIdx.x * 256 + threadIdx.x;
    float t[6] = {0, 0, 0, 0, 0, 0};
    if (i < n) {
        float r1 = rho1[i];
        float s1 = ((r1 > 15.f) ? r1 : log1pf(__expf(r1))) + 1e-5f;
        float ee1 = e1[i];
        float w1 = mu1[i] + ee1 * s1;
        t[0] = -0.5f * ee1 * ee1 - logf(s1) - HL2PI;
        t[1] = -0.5f * w1 * w1 - HL2PI;
        float z1 = w1 * 1000.0f;
        t[2] = -0.5f * z1 * z1 - logf(0.001f) - HL2PI;

        float r2 = rho2[i];
        float s2 = ((r2 > 15.f) ? r2 : log1pf(__expf(r2))) + 1e-5f;
        float ee2 = e2[i];
        float w2 = mu2[i] + ee2 * s2;
        t[3] = -0.5f * ee2 * ee2 - logf(s2) - HL2PI;
        t[4] = -0.5f * w2 * w2 - HL2PI;
        float z2 = w2 * 1000.0f;
        t[5] = -0.5f * z2 * z2 - logf(0.001f) - HL2PI;

        bc[i] = w1 + w2;
    }
    #pragma unroll
    for (int k = 0; k < 6; ++k)
        for (int o = 16; o > 0; o >>= 1) t[k] += __shfl_down(t[k], o);
    __shared__ float sw[8][6];
    int wave = threadIdx.x >> 5;
    if ((threadIdx.x & 31) == 0) {
        #pragma unroll
        for (int k = 0; k < 6; ++k) sw[wave][k] = t[k];
    }
    __syncthreads();
    if (threadIdx.x == 0) {
        #pragma unroll
        for (int k = 0; k < 6; ++k) {
            float a = 0;
            #pragma unroll
            for (int w = 0; w < 8; ++w) a += sw[w][k];
            partial[blockIdx.x * 6 + k] = a;
        }
    }
}

// ---------------- deterministic KL finalize ----------------
__device__ __forceinline__ float block_sum256(float v) {
    __shared__ float s[256];
    s[threadIdx.x] = v;
    __syncthreads();
    for (int o = 128; o > 0; o >>= 1) {
        if (threadIdx.x < o) s[threadIdx.x] += s[threadIdx.x + o];
        __syncthreads();
    }
    float r = s[0];
    __syncthreads();
    return r;
}

__global__ __launch_bounds__(256) void kl_finalize_kernel(const float* __restrict__ pwih, int nwih,
                                                          const float* __restrict__ pwhh, int nwhh,
                                                          const float* __restrict__ pb, int nb,
                                                          float* __restrict__ klout) {
    float sums[12];
    #pragma unroll
    for (int j = 0; j < 3; ++j) {
        float v = 0;
        for (int i = threadIdx.x; i < nwih; i += 256) v += pwih[i * 3 + j];
        sums[j] = block_sum256(v);
    }
    #pragma unroll
    for (int j = 0; j < 3; ++j) {
        float v = 0;
        for (int i = threadIdx.x; i < nwhh; i += 256) v += pwhh[i * 3 + j];
        sums[3 + j] = block_sum256(v);
    }
    #pragma unroll
    for (int j = 0; j < 6; ++j) {
        float v = 0;
        for (int i = threadIdx.x; i < nb; i += 256) v += pb[i * 6 + j];
        sums[6 + j] = block_sum256(v);
    }
    if (threadIdx.x == 0) {
        const float LPI = logf(0.75f), LQ = logf(0.25f);
        float kl = 0.f;
        #pragma unroll
        for (int t = 0; t < 4; ++t) {
            float post = sums[3 * t + 0];
            float a = sums[3 * t + 1] + LPI;
            float b = sums[3 * t + 2] + LQ;
            float mx = fmaxf(a, b);
            float lpr = mx + logf(__expf(a - mx) + __expf(b - mx));
            kl += post - lpr;
        }
        *klout = kl;
    }
}

// ---------------- fused LSTM step ----------------
// gates = [x_t | h] @ [w_ih ; w_hh]^T + bc, then LSTM state update.
// Grid: 512 blocks (8 m-tiles x 64 n-tiles), 256 threads = 8 wave32s.
// Each wave takes 1/8 of K (x: 32, h: 128). Each operand stream uses ONE runtime
// base pointer; every load in the unrolled K loop is base + compile-time constant
// (gate stride 2MB / 512KB, K stride 64B — all fit the signed 24-bit IOFFSET), so
// the loop body is pure load-clauses + WMMAs with no address VALU (no WAR nops).
__global__ __launch_bounds__(256) void lstm_step_kernel(const __bf16* __restrict__ xt,     // [B,I]
                                                        const __bf16* __restrict__ hprev,  // [B,H]
                                                        const __bf16* __restrict__ w_ih,   // [G,I]
                                                        const __bf16* __restrict__ w_hh,   // [G,H]
                                                        const float*  __restrict__ bc,     // [G]
                                                        float*  __restrict__ c_state,      // [B,H]
                                                        __bf16* __restrict__ hnext,        // [B,H]
                                                        float*  __restrict__ out_t,        // [B,H]
                                                        float*  __restrict__ hn_out,       // [B,H]
                                                        float*  __restrict__ cn_out) {     // [B,H]
    __shared__ float lds[4][8][16][16];   // gate, wave, m, n  (32 KB)
    const int tid  = threadIdx.x;
    const int lane = tid & 31;
    const int wave = tid >> 5;
    const int m0 = (blockIdx.x & 7) * 16;
    const int n0 = (blockIdx.x >> 3) * 16;
    const int arow = lane & 15;
    const int aks  = (lane >> 4) * 16;    // A: lanes 0-15 K 0..15, lanes 16-31 K 16..31
    const int bcol = lane & 15;
    const int bks  = (lane >> 4) * 16;    // B: lanes 0-15 K 0..15, lanes 16-31 K 16..31

    // One base pointer per operand stream; all loop offsets become immediates.
    const __bf16* abaseX = xt    + (m0 + arow) * II + aks + wave * 32;
    const __bf16* bbaseX = w_ih  + (n0 + bcol) * II + bks + wave * 32;
    const __bf16* abaseH = hprev + (m0 + arow) * HH + aks + wave * 128;
    const __bf16* bbaseH = w_hh  + (n0 + bcol) * HH + bks + wave * 128;

    v8f acc[4] = {};

    // ---- x part: this wave's K slice [wave*32, wave*32+32) of I=256 ----
    {
        ABv a;
        a.u[0] = *(const u32x4*)(abaseX);
        a.u[1] = *(const u32x4*)(abaseX + 8);
        ABv b[4];
        #pragma unroll
        for (int gi = 0; gi < 4; ++gi) {
            b[gi].u[0] = *(const u32x4*)(bbaseX + gi * (HH * II));       // +512KB*gi
            b[gi].u[1] = *(const u32x4*)(bbaseX + gi * (HH * II) + 8);
        }
        #pragma unroll
        for (int gi = 0; gi < 4; ++gi)
            acc[gi] = __builtin_amdgcn_wmma_f32_16x16x32_bf16(false, a.v, false, b[gi].v,
                                                              (short)0, acc[gi], false, false);
    }
    // ---- h part: this wave's K slice [wave*128, wave*128+128) of H=1024 ----
    #pragma unroll
    for (int kk = 0; kk < 4; ++kk) {
        ABv a;
        a.u[0] = *(const u32x4*)(abaseH + kk * 32);
        a.u[1] = *(const u32x4*)(abaseH + kk * 32 + 8);
        ABv b[4];
        #pragma unroll
        for (int gi = 0; gi < 4; ++gi) {
            b[gi].u[0] = *(const u32x4*)(bbaseH + gi * (HH * HH) + kk * 32);      // +2MB*gi
            b[gi].u[1] = *(const u32x4*)(bbaseH + gi * (HH * HH) + kk * 32 + 8);
        }
        #pragma unroll
        for (int gi = 0; gi < 4; ++gi)
            acc[gi] = __builtin_amdgcn_wmma_f32_16x16x32_bf16(false, a.v, false, b[gi].v,
                                                              (short)0, acc[gi], false, false);
    }

    // ---- stash per-wave partials: C layout VGPR r -> (m = r + 8*(lane>>4), n = lane&15) ----
    #pragma unroll
    for (int gi = 0; gi < 4; ++gi) {
        #pragma unroll
        for (int r = 0; r < 8; ++r) {
            lds[gi][wave][r + 8 * (lane >> 4)][lane & 15] = acc[gi][r];
        }
    }
    __syncthreads();

    // ---- epilogue: thread tid owns (m,n); reduce 8 waves, add bias, LSTM update ----
    const int m = tid >> 4, n = tid & 15;
    const int b = m0 + m, hc = n0 + n;
    float g4[4];
    #pragma unroll
    for (int gi = 0; gi < 4; ++gi) {
        float s = bc[gi * HH + hc];
        #pragma unroll
        for (int w = 0; w < 8; ++w) s += lds[gi][w][m][n];
        g4[gi] = s;
    }
    const float cp = c_state[b * HH + hc];
    const float ig = sigmoidf_(g4[0]);
    const float fg = sigmoidf_(g4[1]);
    const float gg = tanhf(g4[2]);
    const float og = sigmoidf_(g4[3]);
    const float cn = fg * cp + ig * gg;
    const float hn = og * tanhf(cn);
    c_state[b * HH + hc] = cn;
    hnext[b * HH + hc]   = (__bf16)hn;
    __builtin_nontemporal_store(hn, &out_t[b * HH + hc]);   // write-once stream: keep L2 for weights
    hn_out[b * HH + hc]  = hn;
    cn_out[b * HH + hc]  = cn;
}

// ---------------- host ----------------
extern "C" void kernel_launch(void* const* d_in, const int* in_sizes, int n_in,
                              void* d_out, int out_size, void* d_ws, size_t ws_size,
                              hipStream_t stream) {
    const float* x        = (const float*)d_in[0];
    const float* w_ih_mu  = (const float*)d_in[1];
    const float* w_ih_rho = (const float*)d_in[2];
    const float* w_hh_mu  = (const float*)d_in[3];
    const float* w_hh_rho = (const float*)d_in[4];
    const float* b_ih_mu  = (const float*)d_in[5];
    const float* b_ih_rho = (const float*)d_in[6];
    const float* b_hh_mu  = (const float*)d_in[7];
    const float* b_hh_rho = (const float*)d_in[8];
    const float* eps_w_ih = (const float*)d_in[9];
    const float* eps_w_hh = (const float*)d_in[10];
    const float* eps_b_ih = (const float*)d_in[11];
    const float* eps_b_hh = (const float*)d_in[12];

    // ---- workspace carve (256B aligned slabs) ----
    char* ws = (char*)d_ws;
    size_t off = 0;
    auto carve = [&](size_t bytes) -> void* {
        void* p = ws + off;
        off += (bytes + 255) & ~(size_t)255;
        return p;
    };
    __bf16* w_ih_b  = (__bf16*)carve((size_t)GG * II * 2);
    __bf16* w_hh_b  = (__bf16*)carve((size_t)GG * HH * 2);
    float*  bc      = (float*) carve((size_t)GG * 4);
    __bf16* x_b     = (__bf16*)carve((size_t)TT * BB * II * 2);
    __bf16* h_buf0  = (__bf16*)carve((size_t)BB * HH * 2);
    __bf16* h_buf1  = (__bf16*)carve((size_t)BB * HH * 2);
    float*  c_state = (float*) carve((size_t)BB * HH * 4);
    const int nblk_wih = (GG * II) / 256;   // 4096
    const int nblk_whh = (GG * HH) / 256;   // 16384
    const int nblk_b   = GG / 256;          // 16
    float* p_wih = (float*)carve((size_t)nblk_wih * 3 * 4);
    float* p_whh = (float*)carve((size_t)nblk_whh * 3 * 4);
    float* p_b   = (float*)carve((size_t)nblk_b * 6 * 4);

    // ---- d_out layout: output[T*B*H] | h_n[B*H] | c_n[B*H] | kl[1] ----
    float* outp   = (float*)d_out;
    float* hn_out = outp + (size_t)TT * BB * HH;
    float* cn_out = hn_out + (size_t)BB * HH;
    float* kl_out = cn_out + (size_t)BB * HH;

    init_state_kernel<<<(BB * HH + 255) / 256, 256, 0, stream>>>(c_state, h_buf0, BB * HH);
    sample_kl_kernel<<<nblk_wih, 256, 0, stream>>>(w_ih_mu, w_ih_rho, eps_w_ih, w_ih_b, p_wih, GG * II);
    sample_kl_kernel<<<nblk_whh, 256, 0, stream>>>(w_hh_mu, w_hh_rho, eps_w_hh, w_hh_b, p_whh, GG * HH);
    sample_bias_kernel<<<nblk_b, 256, 0, stream>>>(b_ih_mu, b_ih_rho, eps_b_ih,
                                                   b_hh_mu, b_hh_rho, eps_b_hh, bc, p_b, GG);
    kl_finalize_kernel<<<1, 256, 0, stream>>>(p_wih, nblk_wih, p_whh, nblk_whh, p_b, nblk_b, kl_out);
    f32_to_bf16_kernel<<<(TT * BB * II) / 256, 256, 0, stream>>>(x, x_b, TT * BB * II);

    for (int t = 0; t < TT; ++t) {
        const __bf16* hp = (t & 1) ? h_buf1 : h_buf0;
        __bf16*       hn = (t & 1) ? h_buf0 : h_buf1;
        lstm_step_kernel<<<512, 256, 0, stream>>>(x_b + (size_t)t * BB * II, hp,
                                                  w_ih_b, w_hh_b, bc, c_state, hn,
                                                  outp + (size_t)t * BB * HH, hn_out, cn_out);
    }
}
